// AdaptiveGGNN_TTE_73589969649939
// MI455X (gfx1250) — matrix operand order repaired and checked by reading, not compiled, not run on hardware
//
#include <hip/hip_runtime.h>
#include <hip/hip_bf16.h>
#include <math.h>

#define NN   100000
#define EE   1600000
#define FF   32
#define HH   128
#define IEE  32
#define DYNN 16
#define BB   64
#define LL   128
#define DIN  160   // H + IE
#define G3   384   // 3*H

typedef __attribute__((ext_vector_type(16))) _Float16 v16h;
typedef __attribute__((ext_vector_type(8)))  _Float16 v8h;
typedef __attribute__((ext_vector_type(8)))  float    v8f;

// ---------------- WMMA fragment helpers (ISA 7.12.2 layouts, wave32) ----------------
// A (16-bit, 16x32): lane L<16 -> M=L, halves {K0..7, K16..23}; lane>=16 -> {K8..15, K24..31}.
// Both runs are contiguous in row-major memory -> two 16-byte vector loads.
__device__ __forceinline__ v16h wmma_load_a_f16(const _Float16* A, int lda, int row0, int k0, int lane) {
  int r  = row0 + (lane & 15);
  int kb = (lane & 16) ? 8 : 0;
  const _Float16* p = A + (size_t)r * lda + k0 + kb;
  v8h lo = *(const v8h*)p;
  v8h hi = *(const v8h*)(p + 16);
  v16h a;
#pragma unroll
  for (int i = 0; i < 8; ++i) { a[i] = lo[i]; a[8 + i] = hi[i]; }
  return a;
}

__device__ __forceinline__ v16h wmma_load_a_f32(const float* A, int lda, int row0, int k0, int lane) {
  int r  = row0 + (lane & 15);
  int kb = (lane & 16) ? 8 : 0;
  const float* p = A + (size_t)r * lda + k0 + kb;
  float4 l0 = *(const float4*)p;
  float4 l1 = *(const float4*)(p + 4);
  float4 h0 = *(const float4*)(p + 16);
  float4 h1 = *(const float4*)(p + 20);
  v16h a;
  a[0]=(_Float16)l0.x; a[1]=(_Float16)l0.y; a[2]=(_Float16)l0.z; a[3]=(_Float16)l0.w;
  a[4]=(_Float16)l1.x; a[5]=(_Float16)l1.y; a[6]=(_Float16)l1.z; a[7]=(_Float16)l1.w;
  a[8]=(_Float16)h0.x; a[9]=(_Float16)h0.y; a[10]=(_Float16)h0.z; a[11]=(_Float16)h0.w;
  a[12]=(_Float16)h1.x; a[13]=(_Float16)h1.y; a[14]=(_Float16)h1.z; a[15]=(_Float16)h1.w;
  return a;
}

// B fragments pre-swizzled to [kstep][ntile][lane][16] halfs: one 32B contiguous load.
__device__ __forceinline__ v16h load_b_packed(const _Float16* Bp, int NT, int kk, int nt, int lane) {
  return *(const v16h*)(Bp + ((((size_t)kk * NT + nt) * 32 + lane) << 4));
}

__device__ __forceinline__ v8f wmma16(v16h a, v16h b, v8f c) {
  return __builtin_amdgcn_wmma_f32_16x16x32_f16(false, a, false, b, (short)0, c, false, false);
}

__device__ __forceinline__ float sigf(float x) { return 1.f / (1.f + expf(-x)); }

// ---------------- weight packing: f32 -> f16 fragment-swizzled B layout ----------------
// dst[((kk*NT+nt)*32+lane)*16+i] = W_T[kk*32 + ((lane&16)?8:0) + (i&7) + ((i&8)?16:0)][nt*16 + (lane&15)]
__device__ __forceinline__ void pack_frag(const float* W, int ldw, bool trans, int NT,
                                          _Float16* dst, int idx) {
  int i    = idx & 15;
  int lane = (idx >> 4) & 31;
  int tile = idx >> 9;
  int nt = tile % NT, kk = tile / NT;
  int k = kk * 32 + ((lane & 16) ? 8 : 0) + (i & 7) + ((i & 8) ? 16 : 0);
  int n = nt * 16 + (lane & 15);
  float v = trans ? W[(size_t)n * ldw + k] : W[(size_t)k * ldw + n];
  dst[idx] = (_Float16)v;
}

__global__ void k_packw(const float* Wp, const float* Wz, const float* Wr, const float* Wh,
                        const float* Wihf, const float* Wihb, const float* Whhf, const float* Whhb,
                        _Float16* wp16, _Float16* wz16, _Float16* wr16, _Float16* wh16,
                        _Float16* wihf16, _Float16* wihb16, _Float16* whhf16, _Float16* whhb16) {
  int stride = gridDim.x * blockDim.x;
  int t0 = blockIdx.x * blockDim.x + threadIdx.x;
  for (int i = t0; i < 1 * 8 * 512; i += stride) pack_frag(Wp, HH, false, 8, wp16, i);
  for (int i = t0; i < 5 * 8 * 512; i += stride) {
    pack_frag(Wz, HH, false, 8, wz16, i);
    pack_frag(Wr, HH, false, 8, wr16, i);
    pack_frag(Wh, HH, false, 8, wh16, i);
  }
  for (int i = t0; i < 5 * 24 * 512; i += stride) {   // Wih [384][160] -> transposed B
    pack_frag(Wihf, DIN, true, 24, wihf16, i);
    pack_frag(Wihb, DIN, true, 24, wihb16, i);
  }
  for (int i = t0; i < 4 * 24 * 512; i += stride) {   // Whh [384][128] -> transposed B
    pack_frag(Whhf, HH, true, 24, whhf16, i);
    pack_frag(Whhb, HH, true, 24, whhb16, i);
  }
}

// ---------------- h = tanh(x0 @ W_proj + b) : [N,32]x[32,128] ----------------
__global__ void __launch_bounds__(256) k_proj(const float* x0, const _Float16* wp16,
                                              const float* bproj, float* h) {
  int lane = threadIdx.x & 31, wave = threadIdx.x >> 5;
  int row0 = blockIdx.x * 16, n0 = wave * 16;
  v16h a = wmma_load_a_f32(x0, FF, row0, 0, lane);
  v16h b = load_b_packed(wp16, 8, 0, wave, lane);
  v8f acc = {};
  acc = wmma16(a, b, acc);
  int col = n0 + (lane & 15), rowb = row0 + ((lane & 16) ? 8 : 0);
  float bv = bproj[col];
#pragma unroll
  for (int r = 0; r < 8; ++r) h[(size_t)(rowb + r) * HH + col] = tanhf(acc[r] + bv);
}

// ---------------- zero + edge scatter-add (L2-resident f32 atomics) ----------------
__global__ void k_zero(float* p, long n) {
  long stride = (long)gridDim.x * blockDim.x;
  for (long i = (long)blockIdx.x * blockDim.x + threadIdx.x; i < n; i += stride) p[i] = 0.f;
}

__global__ void __launch_bounds__(256) k_scatter(const int* ei, const float* h, float* hagg) {
  int e = blockIdx.x * 8 + (threadIdx.x >> 5);
  if (e >= EE) return;
  int lane = threadIdx.x & 31;
  int src = ei[e], dst = ei[EE + e];
  const float4 v = *(const float4*)(h + (size_t)src * HH + lane * 4);
  float* p = hagg + (size_t)dst * HH + lane * 4;
  atomicAdd(p + 0, v.x); atomicAdd(p + 1, v.y);
  atomicAdd(p + 2, v.z); atomicAdd(p + 3, v.w);
}

// ---------------- a16 = f16(concat[x0, h_agg]) : [N,160] ----------------
__global__ void k_pack_xh(const float* x0, const float* hagg, _Float16* a16) {
  long stride = (long)gridDim.x * blockDim.x;
  for (long idx = (long)blockIdx.x * blockDim.x + threadIdx.x; idx < (long)NN * DIN; idx += stride) {
    int row = (int)(idx / DIN), c = (int)(idx - (long)row * DIN);
    float v = (c < FF) ? x0[(size_t)row * FF + c] : hagg[(size_t)row * HH + (c - FF)];
    a16[idx] = (_Float16)v;
  }
}

// ---------------- z,r gates: 32-row blocks, B fragments reused across 2 M-tiles --------
__global__ void __launch_bounds__(256) k_gates(const _Float16* wz16, const _Float16* wr16,
                                               const float* bz, const float* br,
                                               const float* hagg, float* zbuf, _Float16* a16) {
  int lane = threadIdx.x & 31, wave = threadIdx.x >> 5;
  int row0 = blockIdx.x * 32, n0 = wave * 16;
  v8f z0 = {}, z1 = {}, r0 = {}, r1 = {};
#pragma unroll
  for (int kk = 0; kk < 5; ++kk) {
    v16h a0 = wmma_load_a_f16(a16, DIN, row0, kk * 32, lane);
    v16h a1 = wmma_load_a_f16(a16, DIN, row0 + 16, kk * 32, lane);
    v16h bzf = load_b_packed(wz16, 8, kk, wave, lane);
    v16h brf = load_b_packed(wr16, 8, kk, wave, lane);
    z0 = wmma16(a0, bzf, z0); z1 = wmma16(a1, bzf, z1);
    r0 = wmma16(a0, brf, r0); r1 = wmma16(a1, brf, r1);
  }
  int col = n0 + (lane & 15), rowb = row0 + ((lane & 16) ? 8 : 0);
  float bzv = bz[col], brv = br[col];
  float zv[16], rv[16];
#pragma unroll
  for (int r = 0; r < 8; ++r) {
    zv[r]     = sigf(z0[r] + bzv); rv[r]     = sigf(r0[r] + brv);
    zv[8 + r] = sigf(z1[r] + bzv); rv[8 + r] = sigf(r1[r] + brv);
  }
  __syncthreads();  // all waves done reading a16 rows before in-place overwrite
#pragma unroll
  for (int r = 0; r < 8; ++r) {
    int row = rowb + r;
    zbuf[(size_t)row * HH + col] = zv[r];
    a16[(size_t)row * DIN + FF + col] = (_Float16)(rv[r] * hagg[(size_t)row * HH + col]);
    int row2 = rowb + 16 + r;
    zbuf[(size_t)row2 * HH + col] = zv[8 + r];
    a16[(size_t)row2 * DIN + FF + col] = (_Float16)(rv[8 + r] * hagg[(size_t)row2 * HH + col]);
  }
}

// ---------------- h_tilde GEMM + gated state update (32-row blocks) ----------------
__global__ void __launch_bounds__(256) k_htilde(const _Float16* wh16, const float* bh,
                                                const _Float16* a16, const float* hagg,
                                                const float* zbuf, float* h) {
  int lane = threadIdx.x & 31, wave = threadIdx.x >> 5;
  int row0 = blockIdx.x * 32, n0 = wave * 16;
  v8f acc0 = {}, acc1 = {};
#pragma unroll
  for (int kk = 0; kk < 5; ++kk) {
    v16h b = load_b_packed(wh16, 8, kk, wave, lane);
    v16h a0 = wmma_load_a_f16(a16, DIN, row0, kk * 32, lane);
    v16h a1 = wmma_load_a_f16(a16, DIN, row0 + 16, kk * 32, lane);
    acc0 = wmma16(a0, b, acc0);
    acc1 = wmma16(a1, b, acc1);
  }
  int col = n0 + (lane & 15), rowb = row0 + ((lane & 16) ? 8 : 0);
  float bhv = bh[col];
#pragma unroll
  for (int r = 0; r < 8; ++r) {
    size_t i0 = (size_t)(rowb + r) * HH + col;
    float ht = tanhf(acc0[r] + bhv);
    float z = zbuf[i0];
    h[i0] = (1.f - z) * hagg[i0] + z * ht;
    size_t i1 = (size_t)(rowb + 16 + r) * HH + col;
    float ht1 = tanhf(acc1[r] + bhv);
    float z1 = zbuf[i1];
    h[i1] = (1.f - z1) * hagg[i1] + z1 * ht1;
  }
}

// ---------------- trajectory gather: rnn_in16[b*L+t] = f16([h[node], emb[node]]) ----------
__global__ void k_gather(const float* h, const float* emb, const int* traj, _Float16* rnn16) {
  int stride = gridDim.x * blockDim.x;
  for (int idx = blockIdx.x * blockDim.x + threadIdx.x; idx < BB * LL * DIN; idx += stride) {
    int row = idx / DIN, c = idx - row * DIN;
    int node = traj[row];
    float v;
    if (c < HH) v = h[(size_t)node * HH + c];
    else        v = (node == 0) ? 0.f : emb[(size_t)node * IEE + (c - HH)];  // padding_idx=0
    rnn16[idx] = (_Float16)v;
  }
}

// ---------------- precompute GRU input gates: GX[dir] = rnn_in @ Wih^T + bih ----------
__global__ void __launch_bounds__(256) k_gx(const _Float16* rnn16, const _Float16* wih_all,
                                            const float* bih_f, const float* bih_b, float* gx_all) {
  int dir = blockIdx.z;
  const _Float16* Wih = wih_all + (size_t)dir * DIN * G3;
  const float* bih = dir ? bih_b : bih_f;
  float* GX = gx_all + (size_t)dir * BB * LL * G3;
  int lane = threadIdx.x & 31, wave = threadIdx.x >> 5;
  int row0 = blockIdx.x * 32;
  int nt = blockIdx.y * 8 + wave;
  v8f acc0 = {}, acc1 = {};
#pragma unroll
  for (int kk = 0; kk < 5; ++kk) {
    v16h b = load_b_packed(Wih, 24, kk, nt, lane);
    v16h a0 = wmma_load_a_f16(rnn16, DIN, row0, kk * 32, lane);
    v16h a1 = wmma_load_a_f16(rnn16, DIN, row0 + 16, kk * 32, lane);
    acc0 = wmma16(a0, b, acc0);
    acc1 = wmma16(a1, b, acc1);
  }
  int col = nt * 16 + (lane & 15), rowb = row0 + ((lane & 16) ? 8 : 0);
  float bv = bih[col];
#pragma unroll
  for (int r = 0; r < 8; ++r) {
    GX[(size_t)(rowb + r) * G3 + col] = acc0[r] + bv;
    GX[(size_t)(rowb + 16 + r) * G3 + col] = acc1[r] + bv;
  }
}

// ---------------- persistent sequential BiGRU: 1 block per direction ----------------
__global__ void __launch_bounds__(1024) k_gruseq(const _Float16* whh_all,
                                                 const float* bhh_f, const float* bhh_b,
                                                 const float* gx_all, const int* lengths,
                                                 float* ghbuf_all, float* hout_all) {
  int dir = blockIdx.x;
  const _Float16* Whh = whh_all + (size_t)dir * HH * G3;
  const float* bhh = dir ? bhh_b : bhh_f;
  const float* GX = gx_all + (size_t)dir * BB * LL * G3;
  float* gh   = ghbuf_all + (size_t)dir * BB * G3;
  float* hout = hout_all + (size_t)dir * BB * HH;
  __shared__ __align__(16) _Float16 hA[BB * HH];  // f16 copy for WMMA A fragments (ds_read_b128)
  __shared__ float hS[BB * HH];                   // f32 master state
  int tid = threadIdx.x, lane = tid & 31, wave = tid >> 5;
  for (int i = tid; i < BB * HH; i += 1024) { hS[i] = 0.f; hA[i] = (_Float16)0.f; }
  __syncthreads();
  for (int s = 0; s < LL; ++s) {
    int t = dir ? (LL - 1 - s) : s;
    // phase 1: gh = h @ Whh^T + bhh  ([64,128]x[128,384]); 96 tiles over 32 waves
#pragma unroll
    for (int i = 0; i < 3; ++i) {
      int tile = wave * 3 + i;
      int m0 = (tile / 24) * 16, nt = tile % 24;
      v8f acc = {};
#pragma unroll
      for (int kk = 0; kk < 4; ++kk) {
        v16h a = wmma_load_a_f16(hA, HH, m0, kk * 32, lane);
        v16h b = load_b_packed(Whh, 24, kk, nt, lane);
        acc = wmma16(a, b, acc);
      }
      int col = nt * 16 + (lane & 15), rowb = m0 + ((lane & 16) ? 8 : 0);
      float bv = bhh[col];
#pragma unroll
      for (int r = 0; r < 8; ++r) gh[(size_t)(rowb + r) * G3 + col] = acc[r] + bv;
    }
    __threadfence();
    __syncthreads();
    // phase 2: gate math + masked state update (PyTorch gate order r,z,n)
    for (int e = tid; e < BB * HH; e += 1024) {
      int b = e >> 7, j = e & 127;
      const float* gx = GX + (size_t)(b * LL + t) * G3;
      float ghr = gh[b * G3 + j], ghz = gh[b * G3 + HH + j], ghn = gh[b * G3 + 2 * HH + j];
      float r_ = sigf(gx[j] + ghr);
      float z_ = sigf(gx[HH + j] + ghz);
      float n_ = tanhf(gx[2 * HH + j] + r_ * ghn);
      float hprev = hS[e];
      float hnew = (1.f - z_) * n_ + z_ * hprev;
      float hv = (t < lengths[b]) ? hnew : hprev;
      hS[e] = hv; hA[e] = (_Float16)hv;
    }
    __syncthreads();
  }
  for (int i = tid; i < BB * HH; i += 1024) hout[i] = hS[i];
}

// ---------------- head: LayerNorm -> fc1 (gelu) -> fc2; one block per batch row ----------
__global__ void __launch_bounds__(128) k_head(const float* hout, const float* dyn,
                                              const float* lng, const float* lnbv,
                                              const float* fc1W, const float* fc1b,
                                              const float* fc2W, const float* fc2b, float* out) {
  __shared__ float sbuf[2 * HH + DYNN];
  __shared__ float red[128];
  int b = blockIdx.x, tid = threadIdx.x;
  float v0 = hout[(size_t)b * HH + tid];
  float v1 = hout[(size_t)(BB + b) * HH + tid];
  if (tid < DYNN) sbuf[2 * HH + tid] = dyn[b * DYNN + tid];
  red[tid] = v0 + v1;
  __syncthreads();
  for (int w = 64; w > 0; w >>= 1) { if (tid < w) red[tid] += red[tid + w]; __syncthreads(); }
  float mu = red[0] / 256.f;
  __syncthreads();
  red[tid] = (v0 - mu) * (v0 - mu) + (v1 - mu) * (v1 - mu);
  __syncthreads();
  for (int w = 64; w > 0; w >>= 1) { if (tid < w) red[tid] += red[tid + w]; __syncthreads(); }
  float inv = rsqrtf(red[0] / 256.f + 1e-5f);
  __syncthreads();
  sbuf[tid]      = (v0 - mu) * inv * lng[tid] + lnbv[tid];
  sbuf[HH + tid] = (v1 - mu) * inv * lng[HH + tid] + lnbv[HH + tid];
  __syncthreads();
  float acc = fc1b[tid];
  for (int k = 0; k < 2 * HH + DYNN; ++k) acc += sbuf[k] * fc1W[k * HH + tid];
  float g = 0.5f * acc * (1.f + erff(acc * 0.70710678118f));  // exact gelu
  red[tid] = g * fc2W[tid];
  __syncthreads();
  for (int w = 64; w > 0; w >>= 1) { if (tid < w) red[tid] += red[tid + w]; __syncthreads(); }
  if (tid == 0) out[b] = red[0] + fc2b[0];
}

// =====================================================================================
extern "C" void kernel_launch(void* const* d_in, const int* in_sizes, int n_in,
                              void* d_out, int out_size, void* d_ws, size_t ws_size,
                              hipStream_t stream) {
  (void)in_sizes; (void)n_in; (void)out_size; (void)ws_size;
  const int*   ei      = (const int*)d_in[0];
  const float* x0      = (const float*)d_in[1];
  const int*   traj    = (const int*)d_in[2];
  const int*   lengths = (const int*)d_in[3];
  const float* dyn     = (const float*)d_in[4];
  const float* id_emb  = (const float*)d_in[5];
  const float* W_proj  = (const float*)d_in[6];
  const float* b_proj  = (const float*)d_in[7];
  const float* Wz = (const float*)d_in[8];   const float* bz = (const float*)d_in[9];
  const float* Wr = (const float*)d_in[10];  const float* br = (const float*)d_in[11];
  const float* Wh = (const float*)d_in[12];  const float* bh = (const float*)d_in[13];
  const float* Wih_f = (const float*)d_in[14]; const float* Whh_f = (const float*)d_in[15];
  const float* bih_f = (const float*)d_in[16]; const float* bhh_f = (const float*)d_in[17];
  const float* Wih_b = (const float*)d_in[18]; const float* Whh_b = (const float*)d_in[19];
  const float* bih_b = (const float*)d_in[20]; const float* bhh_b = (const float*)d_in[21];
  const float* ln_g  = (const float*)d_in[22]; const float* ln_b  = (const float*)d_in[23];
  const float* fc1_W = (const float*)d_in[24]; const float* fc1_b = (const float*)d_in[25];
  const float* fc2_W = (const float*)d_in[26]; const float* fc2_b = (const float*)d_in[27];
  float* out = (float*)d_out;

  char* ws = (char*)d_ws;
  size_t off = 0;
  auto walloc = [&](size_t bytes) -> void* {
    void* p = ws + off;
    off += (bytes + 255) & ~(size_t)255;
    return p;
  };
  float*    h     = (float*)   walloc((size_t)NN * HH * 4);
  float*    hagg  = (float*)   walloc((size_t)NN * HH * 4);
  float*    zbuf  = (float*)   walloc((size_t)NN * HH * 4);
  _Float16* a16   = (_Float16*)walloc((size_t)NN * DIN * 2);
  _Float16* wp16  = (_Float16*)walloc((size_t)FF * HH * 2);
  _Float16* wz16  = (_Float16*)walloc((size_t)DIN * HH * 2);
  _Float16* wr16  = (_Float16*)walloc((size_t)DIN * HH * 2);
  _Float16* wh16  = (_Float16*)walloc((size_t)DIN * HH * 2);
  _Float16* wih16 = (_Float16*)walloc((size_t)2 * DIN * G3 * 2);
  _Float16* whh16 = (_Float16*)walloc((size_t)2 * HH * G3 * 2);
  _Float16* rnn16 = (_Float16*)walloc((size_t)BB * LL * DIN * 2);
  float*    gxbuf = (float*)   walloc((size_t)2 * BB * LL * G3 * 4);
  float*    ghbuf = (float*)   walloc((size_t)2 * BB * G3 * 4);
  float*    hout  = (float*)   walloc((size_t)2 * BB * HH * 4);

  k_packw<<<64, 256, 0, stream>>>(W_proj, Wz, Wr, Wh, Wih_f, Wih_b, Whh_f, Whh_b,
                                  wp16, wz16, wr16, wh16,
                                  wih16, wih16 + (size_t)DIN * G3,
                                  whh16, whh16 + (size_t)HH * G3);
  k_proj<<<NN / 16, 256, 0, stream>>>(x0, wp16, b_proj, h);

  for (int s = 0; s < 3; ++s) {
    k_zero<<<8192, 256, 0, stream>>>(hagg, (long)NN * HH);
    k_scatter<<<EE / 8, 256, 0, stream>>>(ei, h, hagg);
    k_pack_xh<<<8192, 256, 0, stream>>>(x0, hagg, a16);
    k_gates<<<NN / 32, 256, 0, stream>>>(wz16, wr16, bz, br, hagg, zbuf, a16);
    k_htilde<<<NN / 32, 256, 0, stream>>>(wh16, bh, a16, hagg, zbuf, h);
  }

  k_gather<<<2048, 256, 0, stream>>>(h, id_emb, traj, rnn16);
  k_gx<<<dim3(BB * LL / 32, 3, 2), 256, 0, stream>>>(rnn16, wih16, bih_f, bih_b, gxbuf);
  k_gruseq<<<2, 1024, 0, stream>>>(whh16, bhh_f, bhh_b, gxbuf, lengths, ghbuf, hout);
  k_head<<<BB, 128, 0, stream>>>(hout, dyn, ln_g, ln_b, fc1_W, fc1_b, fc2_W, fc2_b, out);
}
